// ARM_28217935134778
// MI455X (gfx1250) — matrix operand order repaired and checked by reading, not compiled, not run on hardware
//
#include <hip/hip_runtime.h>

// ---------------- CDNA5 WMMA types ----------------
typedef _Float16 v16h __attribute__((ext_vector_type(16)));
typedef _Float16 v8h  __attribute__((ext_vector_type(8)));
typedef float    v8f  __attribute__((ext_vector_type(8)));

// ---------------- problem constants ----------------
#define IMG_H   256
#define IMG_W   256
#define CHN     30
#define NPIX    (IMG_H * IMG_W)          // 65536
#define IN_DIM  1470                     // 30*24 causal + 750 ups
#define IN_PAD  1472                     // padded to multiple of 32
#define KT1     46                       // 1472 / 32 k-steps for GEMM1
#define HID     2048
#define NBLK    8                        // hidden processed in 8 blocks of 256
#define BM      64                       // pixels per workgroup
#define LDA     1480                     // feat LDS row stride (halves), bank-spread
#define LDH     264                      // h LDS row stride (halves), bank-spread

// LDS carve (dynamic shared)
#define SMEM_H_OFF    (BM * LDA * 2)                  // 189440
#define SMEM_OUT_OFF  (SMEM_H_OFF + BM * LDH * 2)     // 223232
#define SMEM_BYTES    (SMEM_OUT_OFF + BM * 6 * 4)     // 224768  (< 320KB WGP LDS)

// workspace layout
#define W1H_ELEMS ((size_t)KT1 * 128 * 512)           // 3,014,656 halves (~6 MB)
#define W2H_ELEMS ((size_t)64 * 512)                  // 32,768 halves (64 KB)

// =====================================================================
// Prep: W1 (1470x2048 f32) -> f16, pre-swizzled into per-lane WMMA
// B-fragment order.  Tile = (ktile, ntile) of 32x16; within a tile:
// lane l (0..31): N = ntile*16 + (l&15), k_local = (half&7) + 8*(l>>4) + 16*(half>>3),
// 16 contiguous halves per lane (two b128 loads in the GEMM).
// =====================================================================
__global__ void ARM_prep_w1(const float* __restrict__ W1,
                            _Float16* __restrict__ W1h) {
    int tid = blockIdx.x * 256 + threadIdx.x;
    if (tid >= (int)W1H_ELEMS) return;
    int tile  = tid >> 9;          // /512
    int r     = tid & 511;
    int lane  = r >> 4;
    int hh    = r & 15;
    int ktile = tile >> 7;         // 0..45
    int ntile = tile & 127;        // 0..127
    int n = ntile * 16 + (lane & 15);
    int k = ktile * 32 + (hh & 7) + ((lane >> 4) << 3) + ((hh >> 3) << 4);
    float v = (k < IN_DIM) ? W1[(size_t)k * HID + n] : 0.0f;
    W1h[tid] = (_Float16)v;
}

// W2 (2048x6 f32) -> f16 B-fragments, columns padded 6 -> 16.
// 64 k-tiles of 32 covering the full hidden dim.
__global__ void ARM_prep_w2(const float* __restrict__ W2,
                            _Float16* __restrict__ W2h) {
    int tid = blockIdx.x * 256 + threadIdx.x;
    if (tid >= (int)W2H_ELEMS) return;
    int tile = tid >> 9;           // 0..63
    int r    = tid & 511;
    int lane = r >> 4;
    int hh   = r & 15;
    int n = lane & 15;
    int k = tile * 32 + (hh & 7) + ((lane >> 4) << 3) + ((hh >> 3) << 4);   // 0..2047
    float v = (n < 6) ? W2[k * 6 + n] : 0.0f;
    W2h[tid] = (_Float16)v;
}

// helper: assemble a v16h A/B fragment from two 16B chunks
static __device__ __forceinline__ v16h frag16(const _Float16* p) {
    v8h lo = *(const v8h*)(p);
    v8h hi = *(const v8h*)(p + 16);
    return __builtin_shufflevector(lo, hi,
              0,1,2,3,4,5,6,7,8,9,10,11,12,13,14,15);
}

// =====================================================================
// Fused im2col + MLP.  One WG = 64 consecutive pixels.
// Phase 1: gather 64 x 1472 f16 feature tile into LDS (once).
// Phase 2: for each of 8 hidden blocks (256 cols):
//   GEMM1 (WMMA f16): each wave -> 32 pixels x 64 hidden over 46 k-steps
//     (2 m-tiles x 4 n-tiles: each B fragment feeds 2 WMMAs)
//   + b1, ReLU -> h tile (f16) in LDS
//   GEMM2 (WMMA f16): waves 0-3 accumulate h @ W2 (6 cols) in registers
// Finalize: + b2, mu/scale postprocess, store (B,HW,3,2) layout.
// =====================================================================
__global__ void __launch_bounds__(256, 1)
ARM_mlp_kernel(const float* __restrict__ x,
               const float* __restrict__ xu,
               const _Float16* __restrict__ W1h,
               const _Float16* __restrict__ W2h,
               const float* __restrict__ b1,
               const float* __restrict__ b2,
               float* __restrict__ out) {
    extern __shared__ char smem[];
    _Float16* featA = (_Float16*)smem;                    // [BM][LDA]
    _Float16* hS    = (_Float16*)(smem + SMEM_H_OFF);     // [BM][LDH]
    float*    outS  = (float*)(smem + SMEM_OUT_OFF);      // [BM][6]

    const int tid  = threadIdx.x;
    const int pix0 = blockIdx.x * BM;

    // ---------------- Phase 1: feature gather (im2col) ----------------
    // feat[f]: f<720  -> x,  c=f/24, pos=f%24          (causal positions)
    //          f<1470 -> xu, c=f/49, pos=f%49 (flat index == f, offset 720)
    //          else   -> 0 (k padding)
    for (int i = tid; i < BM * IN_PAD; i += 256) {
        int p = i / IN_PAD;
        int f = i - p * IN_PAD;
        float val = 0.0f;
        const float* src = nullptr;
        int c = 0, ky = 0, kx = 0;
        if (f < 720) {
            c = f / 24; int rr = f - c * 24; ky = rr / 7; kx = rr - ky * 7;
            src = x;
        } else if (f < IN_DIM) {
            c = f / 49; int rr = f - c * 49; ky = rr / 7; kx = rr - ky * 7;
            src = xu;
        }
        if (src) {
            int P  = pix0 + p;
            int hr = P >> 8;
            int wc = P & 255;
            int yy = hr + ky - 3;
            int xx = wc + kx - 3;
            if ((unsigned)yy < (unsigned)IMG_H && (unsigned)xx < (unsigned)IMG_W)
                val = src[(size_t)c * NPIX + yy * IMG_W + xx];
        }
        featA[p * LDA + f] = (_Float16)val;
    }
    __syncthreads();

    // ---------------- Phase 2: WMMA GEMMs ----------------
    const int lane   = tid & 31;
    const int wv     = __builtin_amdgcn_readfirstlane(tid >> 5); // wave-uniform
    const int wave_m = wv & 1;       // pixel tile pair: {0,1} or {2,3}
    const int wave_n = wv >> 1;      // hidden quarter 0..3 (64 cols each)
    const int m0     = wave_m * 32;
    const int lcol   = lane & 15;
    const int lhalf  = lane >> 4;    // 0/1: selects k-offset of A/B fragments

    v8f acc2 = {0.f,0.f,0.f,0.f,0.f,0.f,0.f,0.f};   // GEMM2 accumulator (waves 0-3)

    for (int nb = 0; nb < NBLK; ++nb) {
        const int nb0 = nb * 256 + wave_n * 64;

        v8f acc[2][4];
        const v8f vzero = {0.f,0.f,0.f,0.f,0.f,0.f,0.f,0.f};
#pragma unroll
        for (int mt = 0; mt < 2; ++mt)
#pragma unroll
            for (int t = 0; t < 4; ++t) acc[mt][t] = vzero;

        // ---- GEMM1: feat(64x1472) @ W1 block, this wave: 32 x 64 ----
        for (int ks = 0; ks < KT1; ++ks) {
            // A fragments (ISA 16-bit A layout: lanes 0-15 K{0..7,16..23},
            // lanes 16-31 K{8..15,24..31})
            v16h a[2];
#pragma unroll
            for (int mt = 0; mt < 2; ++mt)
                a[mt] = frag16(featA + (m0 + mt * 16 + lcol) * LDA
                               + ks * 32 + lhalf * 8);

            if (ks + 1 < KT1) {  // pull next B tiles into WGP$ (locality 3)
                __builtin_prefetch(W1h + ((size_t)(ks + 1) * 128 +
                                   nb * 16 + wave_n * 4) * 512 + lane * 16, 0, 3);
            }

            // B fragments: 4 n-tiles, each reused by 2 WMMAs
            v16h b[4];
#pragma unroll
            for (int t = 0; t < 4; ++t) {
                int tile = ks * 128 + (nb * 16 + wave_n * 4 + t);
                b[t] = *(const v16h*)(W1h + (size_t)tile * 512 + lane * 16);
            }
#pragma unroll
            for (int mt = 0; mt < 2; ++mt)
#pragma unroll
                for (int t = 0; t < 4; ++t)
                    acc[mt][t] = __builtin_amdgcn_wmma_f32_16x16x32_f16(
                                    false, a[mt], false, b[t], (short)0,
                                    acc[mt][t], false, false);
        }

        // ---- bias + ReLU -> h tile (f16) in LDS ----
#pragma unroll
        for (int t = 0; t < 4; ++t) {
            int n    = nb0 + t * 16 + lcol;
            float bv = b1[n];
            int ncol = wave_n * 64 + t * 16 + lcol;
#pragma unroll
            for (int mt = 0; mt < 2; ++mt) {
#pragma unroll
                for (int j = 0; j < 8; ++j) {
                    int m = m0 + mt * 16 + j + lhalf * 8;   // C/D layout: VGPR j -> row
                    float hv = acc[mt][t][j] + bv;
                    hv = fmaxf(hv, 0.0f);
                    hS[m * LDH + ncol] = (_Float16)hv;
                }
            }
        }
        __syncthreads();

        // ---- GEMM2: h(64x256) @ W2 rows [nb*256 .. nb*256+255] ----
        if (wv < 4) {
#pragma unroll
            for (int ks = 0; ks < 8; ++ks) {
                v16h a = frag16(hS + (wv * 16 + lcol) * LDH + ks * 32 + lhalf * 8);
                v16h b = *(const v16h*)(W2h + (size_t)(nb * 8 + ks) * 512 + lane * 16);
                acc2 = __builtin_amdgcn_wmma_f32_16x16x32_f16(
                          false, a, false, b, (short)0, acc2, false, false);
            }
        }
        __syncthreads();   // hS will be overwritten next block
    }

    // ---------------- Finalize: +b2, postprocess, store ----------------
    if (wv < 4) {
        if (lcol < 6) {
            float bv = b2[lcol];
#pragma unroll
            for (int j = 0; j < 8; ++j) {
                int m = wv * 16 + j + lhalf * 8;
                outS[m * 6 + lcol] = acc2[j] + bv;
            }
        }
    }
    __syncthreads();

    for (int t = tid; t < BM * 6; t += 256) {
        int p   = t / 6;
        int col = t - p * 6;
        float raw = outS[t];
        float o;
        if ((col & 1) == 0) {
            o = raw * 255.0f;                                  // mu
        } else {
            float s = raw;                                     // scale
            o = fminf(fmaxf(s / (1.0f - s + 0.1f) * 100.0f, 1e-8f), 1000.0f);
        }
        out[(size_t)(pix0 + p) * 6 + col] = o;                 // (B,HW,3,2) flat
    }
}

// =====================================================================
extern "C" void kernel_launch(void* const* d_in, const int* in_sizes, int n_in,
                              void* d_out, int out_size, void* d_ws, size_t ws_size,
                              hipStream_t stream) {
    const float* x  = (const float*)d_in[0];   // (1,30,256,256)
    const float* xu = (const float*)d_in[1];   // (1,30,256,256)
    const float* W1 = (const float*)d_in[2];   // (1470,2048)
    const float* b1 = (const float*)d_in[3];   // (2048)
    const float* W2 = (const float*)d_in[4];   // (2048,6)
    const float* b2 = (const float*)d_in[5];   // (6)
    float* out = (float*)d_out;                // (1,65536,3,2)

    _Float16* W1h = (_Float16*)d_ws;
    _Float16* W2h = (_Float16*)((char*)d_ws + W1H_ELEMS * sizeof(_Float16));

    {   // W1 -> swizzled f16 fragments
        int total = (int)W1H_ELEMS;
        ARM_prep_w1<<<(total + 255) / 256, 256, 0, stream>>>(W1, W1h);
    }
    {   // W2 -> swizzled f16 fragments
        int total = (int)W2H_ELEMS;
        ARM_prep_w2<<<(total + 255) / 256, 256, 0, stream>>>(W2, W2h);
    }

    (void)hipFuncSetAttribute((const void*)ARM_mlp_kernel,
                              hipFuncAttributeMaxDynamicSharedMemorySize,
                              SMEM_BYTES);

    ARM_mlp_kernel<<<NPIX / BM, 256, SMEM_BYTES, stream>>>(
        x, xu, W1h, W2h, b1, b2, out);
}
